// KMeansMultiVectorQuantizer_52123723105003
// MI455X (gfx1250) — matrix operand order repaired and checked by reading, not compiled, not run on hardware
//
#include <hip/hip_runtime.h>
#include <hip/hip_bf16.h>

typedef __attribute__((ext_vector_type(16))) _Float16 v16h;
typedef __attribute__((ext_vector_type(8)))  float    v8f;

#define G      8
#define KCODES 1024
#define DG     64
#define BATCH  16
#define TLEN   1024
#define NPOS   (BATCH * TLEN)   /* 16384 */
#define DALL   512
#define PTILES 4                /* position tiles (of 16) per wave */

union V16H { v16h v; uint4 q[2]; };
union F4x2 { float f[8]; float4 q[2]; };
union H2   { _Float16 h[2]; unsigned int u; };
union H8   { _Float16 h[8]; uint4 q; };

// ---------------------------------------------------------------------------
// Kernel 1: codebook -> f16 copy + per-code squared norm. One wave per code row.
// ---------------------------------------------------------------------------
__global__ void vq_prep_embed(const float* __restrict__ embed,
                              _Float16* __restrict__ ecvt,
                              float* __restrict__ esq) {
    int tid  = blockIdx.x * blockDim.x + threadIdx.x;
    int row  = tid >> 5;          // g*KCODES + k, 8192 rows
    int lane = tid & 31;
    const float* src = embed + (size_t)row * DG + 2 * lane;
    float e0 = src[0], e1 = src[1];
    H2 h; h.h[0] = (_Float16)e0; h.h[1] = (_Float16)e1;
    *(unsigned int*)(ecvt + (size_t)row * DG + 2 * lane) = h.u;
    float s = e0 * e0 + e1 * e1;
#pragma unroll
    for (int off = 16; off >= 1; off >>= 1) s += __shfl_xor(s, off, 32);
    if (lane == 0) esq[row] = s;
}

// ---------------------------------------------------------------------------
// Kernel 2: transpose inputs (B, D, T) f32 -> xT[g][n][dg] f16, n = b*T + t.
// Reads coalesced along t; each thread writes 128 contiguous bytes.
// ---------------------------------------------------------------------------
__global__ void vq_prep_x(const float* __restrict__ inputs,
                          _Float16* __restrict__ xT) {
    int tid = blockIdx.x * blockDim.x + threadIdx.x;   // [0, G*NPOS)
    int g = tid / NPOS;
    int n = tid % NPOS;
    int b = n / TLEN, t = n % TLEN;
    const float* src = inputs + ((size_t)b * DALL + g * DG) * TLEN + t;
    _Float16*    dst = xT + ((size_t)g * NPOS + n) * DG;
#pragma unroll
    for (int d0 = 0; d0 < DG; d0 += 8) {
        H8 h;
#pragma unroll
        for (int j = 0; j < 8; ++j)
            h.h[j] = (_Float16)src[(size_t)(d0 + j) * TLEN];
        *(uint4*)(dst + d0) = h.q;
    }
}

// ---------------------------------------------------------------------------
// Kernel 3: distances + argmin. One wave per (group, 64 positions).
// 4 B-tiles (positions) stay register-resident for the whole K-sweep; each
// A-tile (codes) load is amortized over 8 WMMAs -> 4x less codebook L2 traffic.
// A = 16 codes x 32 (K-chunk), B = 32 x 16 positions, C = scores (codes x pos).
// VGPR packing: A and B both hold contiguous K = 16*hi + {0..15} per 32-chunk.
// ---------------------------------------------------------------------------
__global__ void vq_dist_argmin(const _Float16* __restrict__ xT,
                               const _Float16* __restrict__ ecvt,
                               const float* __restrict__ esq,
                               int* __restrict__ idxb) {
    int wave = (blockIdx.x * blockDim.x + threadIdx.x) >> 5;  // 2048 waves
    int lane = threadIdx.x & 31;
    int g  = wave >> 8;           // NPOS/64 = 256 blocks-of-64 per group
    int nb = wave & 255;
    int pl = lane & 15;           // position within tile / code row for A
    int hi = lane >> 4;           // half-wave select
    int p0 = nb * (16 * PTILES) + pl;

    // B tiles: lane holds K = 16*hi + {0..15} of each 32-chunk, column N = pl
    V16H bt[PTILES][2];
#pragma unroll
    for (int j = 0; j < PTILES; ++j) {
        const _Float16* xb = xT + ((size_t)g * NPOS + p0 + j * 16) * DG + 16 * hi;
        bt[j][0].q[0] = *(const uint4*)(xb + 0);
        bt[j][0].q[1] = *(const uint4*)(xb + 8);
        bt[j][1].q[0] = *(const uint4*)(xb + 32);
        bt[j][1].q[1] = *(const uint4*)(xb + 40);
    }

    const int lo = 8 * hi;        // C-side code sub-offset; A-side K sub-offset is 2*lo
    float best[PTILES];
    int   bidx[PTILES];
#pragma unroll
    for (int j = 0; j < PTILES; ++j) { best[j] = __builtin_inff(); bidx[j] = 0; }

    for (int ct = 0; ct < KCODES / 16; ++ct) {
        // A tile: lane holds code row (ct*16 + pl), K = 16*hi + {0..15}/chunk
        const _Float16* ea = ecvt + ((size_t)g * KCODES + ct * 16 + pl) * DG;
        V16H a0, a1;
        a0.q[0] = *(const uint4*)(ea + 2 * lo);
        a0.q[1] = *(const uint4*)(ea + 2 * lo + 8);
        a1.q[0] = *(const uint4*)(ea + 32 + 2 * lo);
        a1.q[1] = *(const uint4*)(ea + 40 + 2 * lo);

        F4x2 eq;
        eq.q[0] = *(const float4*)(esq + (size_t)g * KCODES + ct * 16 + lo);
        eq.q[1] = *(const float4*)(esq + (size_t)g * KCODES + ct * 16 + lo + 4);

#pragma unroll
        for (int j = 0; j < PTILES; ++j) {
            v8f acc = {};
            acc = __builtin_amdgcn_wmma_f32_16x16x32_f16(false, a0.v, false, bt[j][0].v,
                                                         (short)0, acc, false, false);
            acc = __builtin_amdgcn_wmma_f32_16x16x32_f16(false, a1.v, false, bt[j][1].v,
                                                         (short)0, acc, false, false);
            // C layout: lane holds column N=pl, codes M = lo + r (r = VGPR index 0..7)
#pragma unroll
            for (int r = 0; r < 8; ++r) {
                float s   = eq.f[r] - 2.0f * acc[r];  // |e|^2 - 2 x.e (|x|^2 irrelevant)
                int  code = ct * 16 + lo + r;
                if (s < best[j]) { best[j] = s; bidx[j] = code; }
            }
        }
    }
    // combine the two half-wave code subsets for each position
#pragma unroll
    for (int j = 0; j < PTILES; ++j) {
        float ob = __shfl_xor(best[j], 16, 32);
        int   oi = __shfl_xor(bidx[j], 16, 32);
        if (ob < best[j] || (ob == best[j] && oi < bidx[j])) { best[j] = ob; bidx[j] = oi; }
        if (hi == 0) idxb[(size_t)g * NPOS + p0 + j * 16] = bidx[j];
    }
}

// ---------------------------------------------------------------------------
// Kernel 4: gather codewords (f32), write z_q, SSE partials (deterministic),
// float histogram (exact integer adds -> order-independent).
// ---------------------------------------------------------------------------
__global__ void vq_gather(const float* __restrict__ inputs,
                          const float* __restrict__ embed,
                          const int* __restrict__ idxb,
                          float* __restrict__ out,
                          float* __restrict__ counts,
                          float* __restrict__ partials) {
    __shared__ float wred[8];
    int tid = blockIdx.x * blockDim.x + threadIdx.x;
    int g = tid / NPOS;
    int n = tid % NPOS;
    int b = n / TLEN, t = n % TLEN;
    int idx = idxb[(size_t)g * NPOS + n];
    const float* e = embed + ((size_t)g * KCODES + idx) * DG;
    const float* x = inputs + ((size_t)b * DALL + g * DG) * TLEN + t;
    float*       o = out    + ((size_t)b * DALL + g * DG) * TLEN + t;
    float sse = 0.f;
#pragma unroll
    for (int d = 0; d < DG; ++d) {
        float ev = e[d];
        float xv = x[(size_t)d * TLEN];
        o[(size_t)d * TLEN] = ev;                  // zq_st forward value == zq
        float df = ev - xv;
        sse += df * df;
    }
    atomicAdd(&counts[(size_t)g * KCODES + idx], 1.0f);
#pragma unroll
    for (int off = 16; off >= 1; off >>= 1) sse += __shfl_xor(sse, off, 32);
    int lane = threadIdx.x & 31;
    int w    = threadIdx.x >> 5;
    if (lane == 0) wred[w] = sse;
    __syncthreads();
    if (threadIdx.x == 0) {
        float s = 0.f;
#pragma unroll
        for (int i = 0; i < 8; ++i) s += wred[i];
        partials[blockIdx.x] = s;                  // fixed order -> deterministic
    }
}

// ---------------------------------------------------------------------------
// Kernel 5: finalize loss / perplexity / kldiv. Single 1024-thread block.
// ---------------------------------------------------------------------------
__global__ void vq_finalize(const float* __restrict__ counts,
                            const float* __restrict__ partials,
                            float* __restrict__ out) {
    __shared__ float sred[1024];
    int tid = threadIdx.x;
    sred[tid] = (tid < 512) ? partials[tid] : 0.f;
    __syncthreads();
    for (int s = 512; s > 0; s >>= 1) {
        if (tid < s) sred[tid] += sred[tid + s];
        __syncthreads();
    }
    float sse = sred[0];
    __syncthreads();
    float perp_sum = 0.f;
    for (int g = 0; g < G; ++g) {
        float c = counts[g * KCODES + tid];
        float p = c * (1.0f / (float)NPOS);
        sred[tid] = p * logf(p + 1e-10f);
        __syncthreads();
        for (int s = 512; s > 0; s >>= 1) {
            if (tid < s) sred[tid] += sred[tid + s];
            __syncthreads();
        }
        if (tid == 0) perp_sum += expf(-sred[0]);
        __syncthreads();
    }
    if (tid == 0) {
        const size_t Z = (size_t)BATCH * DALL * TLEN;   // 8388608
        // vq_loss == commit_loss numerically -> loss = 1.25 * mse
        out[Z] = 1.25f * sse / (float)((size_t)G * NPOS * DG);
        float kld = (float)G * logf((float)KCODES) * (float)TLEN;
        for (int i = 0; i < BATCH; ++i) out[Z + 1 + i] = kld;
        out[Z + 1 + BATCH] = perp_sum / (float)G;
    }
}

// ---------------------------------------------------------------------------
extern "C" void kernel_launch(void* const* d_in, const int* in_sizes, int n_in,
                              void* d_out, int out_size, void* d_ws, size_t ws_size,
                              hipStream_t stream) {
    (void)in_sizes; (void)n_in; (void)out_size; (void)ws_size;
    const float* inputs = (const float*)d_in[0];   // (16, 512, 1024) f32
    const float* embed  = (const float*)d_in[1];   // (8, 1024, 64)  f32
    float* out = (float*)d_out;

    char* w = (char*)d_ws;
    size_t off = 0;
    _Float16* ecvt = (_Float16*)(w + off); off += (size_t)G * KCODES * DG * 2; // 1 MiB
    float*    esq  = (float*)(w + off);    off += (size_t)G * KCODES * 4;      // 32 KiB
    _Float16* xT   = (_Float16*)(w + off); off += (size_t)G * NPOS * DG * 2;   // 16 MiB
    int*      idxb = (int*)(w + off);      off += (size_t)G * NPOS * 4;        // 512 KiB
    float*  counts = (float*)(w + off);    off += (size_t)G * KCODES * 4;      // 32 KiB
    float* partials= (float*)(w + off);    off += 512 * 4;

    hipMemsetAsync(counts, 0, (size_t)G * KCODES * 4, stream);     // re-zero every call
    vq_prep_embed <<<1024, 256, 0, stream>>>(embed, ecvt, esq);    // 8192 rows * 1 wave
    vq_prep_x     <<< 512, 256, 0, stream>>>(inputs, xT);          // 131072 threads
    vq_dist_argmin<<< 256, 256, 0, stream>>>(xT, ecvt, esq, idxb); // 2048 waves x 64 pos
    vq_gather     <<< 512, 256, 0, stream>>>(inputs, embed, idxb, out, counts, partials);
    vq_finalize   <<<   1, 1024, 0, stream>>>(counts, partials, out);
}
